// TransformerEncoderLayer_18442589569162
// MI455X (gfx1250) — matrix operand order repaired and checked by reading, not compiled, not run on hardware
//
#include <hip/hip_runtime.h>
#include <math.h>

// ---------------------------------------------------------------------------
// CDNA5 (gfx1250) T5 encoder layer.
//  - all GEMMs: TDM (tensor_load_to_lds) double-buffered LDS staging feeding
//    v_wmma_f32_16x16x32_bf16
//  - attention: per-wave flash-style two-pass softmax with WMMA score/PV GEMMs
// B=4 L=1024 D=1024 H=16 DK=64 DFF=4096, NUM_BUCKETS=32 MAX_DIST=128
// ---------------------------------------------------------------------------

typedef __attribute__((ext_vector_type(16))) __bf16 v16bf;
typedef __attribute__((ext_vector_type(8)))  __bf16 v8bf;
typedef __attribute__((ext_vector_type(8)))  float  v8f;
typedef __attribute__((ext_vector_type(4)))  unsigned int v4u;
typedef __attribute__((ext_vector_type(8)))  int v8i;
typedef __attribute__((ext_vector_type(4)))  int v4i;

#define WMMA_BF16(a, b, c) \
  __builtin_amdgcn_wmma_f32_16x16x32_bf16(false, (a), false, (b), (short)0, (c), false, false)

// --- fragment loaders (contiguous k order, shared by A and B) -----------------------
// A-matrix 16x32 bf16: lane l -> row (l&15); contiguous K-run of 16 at c0=(l>>4)*16
__device__ __forceinline__ v16bf load_afrag(const __bf16* p, int ld) {
  const int lane = threadIdx.x & 31;
  const int r  = lane & 15;
  const int c0 = (lane >> 4) * 16;
  const __bf16* q = p + (size_t)r * ld + c0;
  v8bf lo = *(const v8bf*)(q);
  v8bf hi = *(const v8bf*)(q + 8);
  v16bf out;
#pragma unroll
  for (int i = 0; i < 8; ++i) { out[i] = lo[i]; out[i + 8] = hi[i]; }
  return out;
}

// B-matrix 32x16 bf16 (= W^T with W row-major [N,K]):
// lane l -> W-row (l&15), contiguous K-run of 16 at (l>>4)*16
__device__ __forceinline__ v16bf load_bfrag(const __bf16* p, int ld) {
  const int lane = threadIdx.x & 31;
  const int n  = lane & 15;
  const int k0 = (lane >> 4) * 16;
  return *(const v16bf*)(p + (size_t)n * ld + k0);
}

// ---------------------------------------------------------------------------
// TDM: issue a 2-D bf16 tile load Global -> LDS (D# per ISA ch.8).
//  group0: count=1 | lds_addr | global_addr(57b) | type=2
//  group1: data_size=1(2B), tensor_dim0/1, tile_dim0/1 (tile_dim2=0),
//          tensor_dim0_stride (stride1=0, unused for 2-D)
//  groups 2/3 + extra group: zero (2-D tile)
// ---------------------------------------------------------------------------
__device__ __forceinline__ void tdm_load_2d(unsigned lds_byte,
                                            unsigned long long gaddr,
                                            unsigned ten_d0, unsigned ten_d1,
                                            unsigned long long stride0,
                                            unsigned tile_d0, unsigned tile_d1) {
  v4u g0;
  g0[0] = 1u;                                                  // count=1 (valid)
  g0[1] = lds_byte;                                            // lds_addr [63:32]
  g0[2] = (unsigned)gaddr;                                     // global_addr lo
  g0[3] = ((unsigned)(gaddr >> 32) & 0x01FFFFFFu) | (2u << 30);// addr[56:32]|type=2
  v8i g1;
  g1[0] = (int)(1u << 16);                                     // data_size=2B
  g1[1] = (int)((ten_d0 & 0xFFFFu) << 16);                     // tensor_dim0 lo16
  g1[2] = (int)((ten_d0 >> 16) | ((ten_d1 & 0xFFFFu) << 16));  // dim0 hi / dim1 lo
  g1[3] = (int)((ten_d1 >> 16) | (tile_d0 << 16));             // dim1 hi / tile0
  g1[4] = (int)tile_d1;                                        // tile1, tile2=0
  g1[5] = (int)(unsigned)stride0;                              // stride0 [31:0]
  g1[6] = (int)((unsigned)(stride0 >> 32) & 0xFFFFu);          // stride0 hi, stride1=0
  g1[7] = 0;
  v4i z4 = {0, 0, 0, 0};
  v8i z8 = {0, 0, 0, 0, 0, 0, 0, 0};
  __builtin_amdgcn_tensor_load_to_lds(g0, g1, z4, z4, z8, 0);
}

// ---------------------------------------------------------------------------
// elementwise: f32 -> bf16 cast
// ---------------------------------------------------------------------------
__global__ void cast_bf16_kernel(const float* __restrict__ x, __bf16* __restrict__ y, int n) {
  int i = blockIdx.x * 256 + threadIdx.x;
  if (i < n) y[i] = (__bf16)x[i];
}

// ---------------------------------------------------------------------------
// T5 layernorm (rms, no mean/bias) -> bf16
// ---------------------------------------------------------------------------
__global__ void t5ln_kernel(const float* __restrict__ x, const float* __restrict__ w,
                            __bf16* __restrict__ y, int D, float eps) {
  const int row = blockIdx.x;
  const float* xr = x + (size_t)row * D;
  float ss = 0.f;
  for (int c = threadIdx.x; c < D; c += 256) { float v = xr[c]; ss += v * v; }
  __shared__ float red[256];
  red[threadIdx.x] = ss;
  __syncthreads();
#pragma unroll
  for (int s = 128; s > 0; s >>= 1) {
    if (threadIdx.x < s) red[threadIdx.x] += red[threadIdx.x + s];
    __syncthreads();
  }
  float inv = rsqrtf(red[0] / (float)D + eps);
  for (int c = threadIdx.x; c < D; c += 256)
    y[(size_t)row * D + c] = (__bf16)(w[c] * xr[c] * inv);
}

// ---------------------------------------------------------------------------
// relative-position bucket table: tab[rel+1023][h] = rel_bias[bucket][h]
// ---------------------------------------------------------------------------
__device__ __forceinline__ int rp_bucket(int rel) {  // rel = mem - ctx
  int n = -rel;
  int ret = (n < 0) ? 16 : 0;
  n = (n < 0) ? -n : n;
  if (n < 8) return ret + n;
  int vil = 8 + (int)(logf((float)n * 0.125f) * 2.8853901817f);  // 8/log(16)
  vil = vil < 15 ? vil : 15;
  return ret + vil;
}

__global__ void bias_tab_kernel(const float* __restrict__ rel_bias, float* __restrict__ tab) {
  int i = blockIdx.x * 256 + threadIdx.x;
  if (i >= 2047 * 16) return;
  int h = i & 15, ridx = i >> 4;
  tab[ridx * 16 + h] = rel_bias[rp_bucket(ridx - 1023) * 16 + h];
}

// broadcast position_bias [B,H,L,L] (float4-vectorized writes, 256MB)
__global__ void posbias_kernel(const int* __restrict__ pos, const float* __restrict__ tab,
                               float* __restrict__ pb) {
  size_t base = ((size_t)blockIdx.x * 256 + threadIdx.x) * 4;  // over 2^26 elems
  int k = (int)(base & 1023);
  int q = (int)((base >> 10) & 1023);
  int h = (int)((base >> 20) & 15);
  int b = (int)(base >> 24);
  int pq = pos[b * 1024 + q];
  const int* pk = pos + b * 1024 + k;
  float4 v;
  v.x = tab[(pk[0] - pq + 1023) * 16 + h];
  v.y = tab[(pk[1] - pq + 1023) * 16 + h];
  v.z = tab[(pk[2] - pq + 1023) * 16 + h];
  v.w = tab[(pk[3] - pq + 1023) * 16 + h];
  *(float4*)(pb + base) = v;
}

// ---------------------------------------------------------------------------
// WMMA GEMM with TDM double-buffered LDS staging.
// C[M,N] = epi(A[M,K] @ W[N,K]^T).  block = 8 waves (2x4), wave tile 64x64,
// block tile 128x256, K-step 32.  Wave 0 drives the TDM; s_wait_tensorcnt(2)
// keeps the next stage's pair of DMAs in flight behind the WMMA stream.
// ---------------------------------------------------------------------------
#define GEMM_BM 128
#define GEMM_BN 256
#define GEMM_KS 32
#define GEMM_ATILE (GEMM_BM * GEMM_KS)           // 4096 bf16 = 8KB
#define GEMM_WTILE (GEMM_BN * GEMM_KS)           // 8192 bf16 = 16KB
#define GEMM_BUF   (GEMM_ATILE + GEMM_WTILE)     // 12288 bf16 = 24KB

template <bool GELU, bool TRANSV, bool OUTF32>
__global__ void __launch_bounds__(256)
gemm_kernel(const __bf16* __restrict__ A, const __bf16* __restrict__ W,
            const float* __restrict__ bias, const float* __restrict__ res, float scale,
            float* __restrict__ outF, __bf16* __restrict__ outB, int K, int N, int Mtot) {
  __shared__ __align__(128) __bf16 smem[2 * GEMM_BUF];  // 48KB

  const int lane = threadIdx.x & 31;
  const int wave = threadIdx.x >> 5;
  const int wm = wave >> 2, wn = wave & 3;
  const int m0b = blockIdx.x * GEMM_BM;
  const int n0b = blockIdx.y * GEMM_BN;

  // LDS byte offset of smem = low 32 bits of its flat address (ISA 10.2:
  // LDS aperture maps addr[31:0] to the wave's LDS space).
  const unsigned lds0 = (unsigned)(unsigned long long)(uintptr_t)smem;
  const unsigned long long gA = (unsigned long long)(uintptr_t)A;
  const unsigned long long gW = (unsigned long long)(uintptr_t)W;

  // scalar (SGPR) branch: TDM ignores EXEC, so gate by wave id, not lanes
  const int uwave = __builtin_amdgcn_readfirstlane(wave);

  auto issue = [&](int kb, int b) {
    tdm_load_2d(lds0 + (unsigned)(b * GEMM_BUF * 2),
                gA + 2ull * ((unsigned long long)m0b * K + kb),
                (unsigned)(K - kb), (unsigned)(Mtot - m0b),
                (unsigned long long)K, GEMM_KS, GEMM_BM);
    tdm_load_2d(lds0 + (unsigned)(b * GEMM_BUF * 2 + GEMM_ATILE * 2),
                gW + 2ull * ((unsigned long long)n0b * K + kb),
                (unsigned)(K - kb), (unsigned)(N - n0b),
                (unsigned long long)K, GEMM_KS, GEMM_BN);
  };

  v8f acc[4][4];
#pragma unroll
  for (int i = 0; i < 4; ++i)
#pragma unroll
    for (int j = 0; j < 4; ++j) acc[i][j] = (v8f){0.f, 0.f, 0.f, 0.f, 0.f, 0.f, 0.f, 0.f};

  const int nsteps = K / GEMM_KS;
  int buf = 0;
  if (uwave == 0) issue(0, 0);

  for (int s = 0; s < nsteps; ++s) {
    if (uwave == 0) {
      if (s + 1 < nsteps) {
        issue((s + 1) * GEMM_KS, buf ^ 1);
        __builtin_amdgcn_s_wait_tensorcnt(2);  // current buffer's pair done
      } else {
        __builtin_amdgcn_s_wait_tensorcnt(0);
      }
    }
    __syncthreads();  // publish LDS tiles to all 8 waves

    const __bf16* Abuf = smem + buf * GEMM_BUF;
    const __bf16* Wbuf = Abuf + GEMM_ATILE;
    v16bf af[4], bf[4];
#pragma unroll
    for (int i = 0; i < 4; ++i)
      af[i] = load_afrag(Abuf + (wm * 64 + 16 * i) * GEMM_KS, GEMM_KS);
#pragma unroll
    for (int j = 0; j < 4; ++j)
      bf[j] = load_bfrag(Wbuf + (wn * 64 + 16 * j) * GEMM_KS, GEMM_KS);
#pragma unroll
    for (int i = 0; i < 4; ++i)
#pragma unroll
      for (int j = 0; j < 4; ++j) acc[i][j] = WMMA_BF16(af[i], bf[j], acc[i][j]);

    __syncthreads();  // reads done before this buffer is DMA-overwritten
    buf ^= 1;
  }

  const int m0 = m0b + wm * 64;
  const int n0 = n0b + wn * 64;
  const int rbase = (lane >> 4) * 8;
  const int cn = lane & 15;
#pragma unroll
  for (int i = 0; i < 4; ++i)
#pragma unroll
    for (int j = 0; j < 4; ++j) {
      const int colg = n0 + 16 * j + cn;
      const float bv = bias ? bias[colg] : 0.f;
#pragma unroll
      for (int r = 0; r < 8; ++r) {
        const int mg = m0 + 16 * i + rbase + r;
        float v = (acc[i][j][r] + bv) * scale;
        if (GELU) v = 0.5f * v * (1.f + erff(v * 0.70710678118f));
        if (res) v += res[(size_t)mg * N + colg];
        if (OUTF32) {
          outF[(size_t)mg * N + colg] = v;
        } else if (TRANSV) {
          // V: [b,l,h,dk] -> Vt [b,h,dk,l]
          const int bb = mg >> 10, ll = mg & 1023, hh = colg >> 6, dd = colg & 63;
          outB[(((size_t)(bb * 16 + hh)) * 64 + dd) * 1024 + ll] = (__bf16)v;
        } else {
          outB[(size_t)mg * N + colg] = (__bf16)v;
        }
      }
    }
}

// ---------------------------------------------------------------------------
// Attention: one wave per (b, h, 16 q-rows).  Two-pass softmax:
// pass A = streaming max/sumexp, pass B = probs -> attn_sum (atomic) + P@V.
// Bias is read back from d_out's position_bias region (L2-warm).
// ---------------------------------------------------------------------------
__global__ void __launch_bounds__(32)
attn_kernel(const __bf16* __restrict__ Q, const __bf16* __restrict__ Kt,
            const __bf16* __restrict__ Vt, const float* __restrict__ pb,
            float* __restrict__ attn_sum, __bf16* __restrict__ ctx) {
  const int lane = threadIdx.x;
  const int qt = blockIdx.x, h = blockIdx.y, b = blockIdx.z;
  const int L = 1024, HD = 1024;
  const int q0 = qt * 16;
  const __bf16* Qp = Q + ((size_t)b * L) * HD + h * 64;
  const __bf16* Kp = Kt + ((size_t)b * L) * HD + h * 64;
  const __bf16* Vp = Vt + ((size_t)(b * 16 + h)) * 64 * L;  // [dk][l]
  const float* pbp = pb + (((size_t)(b * 16 + h)) * L + q0) * L;

  const v16bf qa0 = load_afrag(Qp + (size_t)q0 * HD, HD);
  const v16bf qa1 = load_afrag(Qp + (size_t)q0 * HD + 32, HD);

  const int rbase = (lane >> 4) * 8;
  const int cn = lane & 15;

  float mrow[8], lrow[8];
#pragma unroll
  for (int r = 0; r < 8; ++r) { mrow[r] = -1e30f; lrow[r] = 0.f; }

  // ---- pass A: streaming row max / sum-exp ----
  for (int kt = 0; kt < L; kt += 16) {
    v16bf bk0 = load_bfrag(Kp + (size_t)kt * HD, HD);
    v16bf bk1 = load_bfrag(Kp + (size_t)kt * HD + 32, HD);
    v8f s = (v8f){0.f, 0.f, 0.f, 0.f, 0.f, 0.f, 0.f, 0.f};
    s = WMMA_BF16(qa0, bk0, s);
    s = WMMA_BF16(qa1, bk1, s);
#pragma unroll
    for (int r = 0; r < 8; ++r) {
      float sv = s[r] + pbp[(size_t)(rbase + r) * L + kt + cn];
      float bm = sv;
#pragma unroll
      for (int off = 1; off < 16; off <<= 1) bm = fmaxf(bm, __shfl_xor(bm, off, 32));
      float nm = fmaxf(mrow[r], bm);
      float ev = __expf(sv - nm);
      float bs = ev;
#pragma unroll
      for (int off = 1; off < 16; off <<= 1) bs += __shfl_xor(bs, off, 32);
      lrow[r] = lrow[r] * __expf(mrow[r] - nm) + bs;
      mrow[r] = nm;
    }
  }
  float rl[8];
#pragma unroll
  for (int r = 0; r < 8; ++r) rl[r] = 1.f / lrow[r];

  // ---- pass B: probs -> attn_sum + O = P @ V ----
  __shared__ __align__(32) __bf16 ptile[16 * 32];
  v8f o[4];
#pragma unroll
  for (int n = 0; n < 4; ++n) o[n] = (v8f){0.f, 0.f, 0.f, 0.f, 0.f, 0.f, 0.f, 0.f};

  for (int kb = 0; kb < L; kb += 32) {
#pragma unroll
    for (int hf = 0; hf < 2; ++hf) {
      const int kt = kb + hf * 16;
      v16bf bk0 = load_bfrag(Kp + (size_t)kt * HD, HD);
      v16bf bk1 = load_bfrag(Kp + (size_t)kt * HD + 32, HD);
      v8f s = (v8f){0.f, 0.f, 0.f, 0.f, 0.f, 0.f, 0.f, 0.f};
      s = WMMA_BF16(qa0, bk0, s);
      s = WMMA_BF16(qa1, bk1, s);
#pragma unroll
      for (int r = 0; r < 8; ++r) {
        const int row = rbase + r;
        float sv = s[r] + pbp[(size_t)row * L + kt + cn];
        float p = __expf(sv - mrow[r]) * rl[r];
        ptile[row * 32 + hf * 16 + cn] = (__bf16)p;
        atomicAdd(&attn_sum[((size_t)b * L + q0 + row) * L + kt + cn], p);
      }
    }
    v16bf pa = load_afrag(ptile, 32);
#pragma unroll
    for (int n = 0; n < 4; ++n) {
      v16bf bv = load_bfrag(Vp + (size_t)(n * 16) * L + kb, L);
      o[n] = WMMA_BF16(pa, bv, o[n]);
    }
  }

  // ctx layout [b, l, h*64 + d] (bf16) feeds the fc GEMM directly
#pragma unroll
  for (int n = 0; n < 4; ++n)
#pragma unroll
    for (int r = 0; r < 8; ++r)
      ctx[((size_t)b * L + q0 + rbase + r) * HD + h * 64 + n * 16 + cn] = (__bf16)o[n][r];
}

// ---------------------------------------------------------------------------
// launch
// ---------------------------------------------------------------------------
extern "C" void kernel_launch(void* const* d_in, const int* in_sizes, int n_in,
                              void* d_out, int out_size, void* d_ws, size_t ws_size,
                              hipStream_t stream) {
  (void)in_sizes; (void)n_in; (void)out_size; (void)ws_size;
  const int Bn = 4, Ln = 1024, Dn = 1024, Hn = 16, DFFn = 4096;
  const int M = Bn * Ln;  // 4096

  const float* src   = (const float*)d_in[0];
  const int*   pos   = (const int*)  d_in[1];
  const float* ln1_w = (const float*)d_in[2];
  const float* wq    = (const float*)d_in[3];
  const float* bq    = (const float*)d_in[4];
  const float* wk    = (const float*)d_in[5];
  const float* bk    = (const float*)d_in[6];
  const float* wv    = (const float*)d_in[7];
  const float* bv    = (const float*)d_in[8];
  const float* fc_w  = (const float*)d_in[9];
  const float* fc_b  = (const float*)d_in[10];
  const float* relb  = (const float*)d_in[11];
  const float* ln2_w = (const float*)d_in[12];
  const float* wi    = (const float*)d_in[13];
  const float* wo    = (const float*)d_in[14];

  float* out_main = (float*)d_out;                    // [B,L,D]
  float* out_asum = out_main + (size_t)M * Dn;        // [B,L,L]
  float* out_pb   = out_asum + (size_t)Bn * Ln * Ln;  // [B,H,L,L]

  // workspace carve-out (256B aligned chunks)
  char* wp = (char*)d_ws;
  auto alloc = [&](size_t bytes) { char* p = wp; wp += (bytes + 255) & ~(size_t)255; return p; };
  __bf16* projbf = (__bf16*)alloc((size_t)M * Dn * 2);
  __bf16* wq_bf  = (__bf16*)alloc((size_t)Dn * Dn * 2);
  __bf16* wk_bf  = (__bf16*)alloc((size_t)Dn * Dn * 2);
  __bf16* wv_bf  = (__bf16*)alloc((size_t)Dn * Dn * 2);
  __bf16* fcw_bf = (__bf16*)alloc((size_t)Dn * Dn * 2);
  __bf16* wi_bf  = (__bf16*)alloc((size_t)DFFn * Dn * 2);
  __bf16* wo_bf  = (__bf16*)alloc((size_t)Dn * DFFn * 2);
  __bf16* Qbf    = (__bf16*)alloc((size_t)M * Dn * 2);
  __bf16* Kbf    = (__bf16*)alloc((size_t)M * Dn * 2);
  __bf16* Vtb    = (__bf16*)alloc((size_t)M * Dn * 2);   // [b,h,dk,l]
  __bf16* ctxbf  = (__bf16*)alloc((size_t)M * Dn * 2);
  float*  attn_o = (float*) alloc((size_t)M * Dn * 4);
  __bf16* nxbf   = (__bf16*)alloc((size_t)M * Dn * 2);
  __bf16* hbf    = (__bf16*)alloc((size_t)M * DFFn * 2);
  float*  tab    = (float*) alloc((size_t)2047 * Hn * 4);

  // attn_sum is accumulated atomically -> zero it every call
  (void)hipMemsetAsync(out_asum, 0, (size_t)Bn * Ln * Ln * sizeof(float), stream);

  // weight casts fp32 -> bf16
  auto cast = [&](const float* x, __bf16* y, int n) {
    cast_bf16_kernel<<<(n + 255) / 256, 256, 0, stream>>>(x, y, n);
  };
  cast(wq, wq_bf, Dn * Dn);
  cast(wk, wk_bf, Dn * Dn);
  cast(wv, wv_bf, Dn * Dn);
  cast(fc_w, fcw_bf, Dn * Dn);
  cast(wi, wi_bf, DFFn * Dn);
  cast(wo, wo_bf, Dn * DFFn);

  // LN1
  t5ln_kernel<<<M, 256, 0, stream>>>(src, ln1_w, projbf, Dn, 1e-12f);

  // position bias: table + broadcast into d_out
  bias_tab_kernel<<<(2047 * 16 + 255) / 256, 256, 0, stream>>>(relb, tab);
  posbias_kernel<<<65536, 256, 0, stream>>>(pos, tab, out_pb);

  // QKV projections (Q folds the 1/sqrt(DK) scale; V written pre-transposed)
  dim3 g1(M / GEMM_BM, Dn / GEMM_BN);
  gemm_kernel<false, false, false><<<g1, 256, 0, stream>>>(projbf, wq_bf, bq, nullptr, 0.125f,
                                                           nullptr, Qbf, Dn, Dn, M);
  gemm_kernel<false, false, false><<<g1, 256, 0, stream>>>(projbf, wk_bf, bk, nullptr, 1.0f,
                                                           nullptr, Kbf, Dn, Dn, M);
  gemm_kernel<false, true, false><<<g1, 256, 0, stream>>>(projbf, wv_bf, bv, nullptr, 1.0f,
                                                          nullptr, Vtb, Dn, Dn, M);

  // attention
  attn_kernel<<<dim3(Ln / 16, Hn, Bn), 32, 0, stream>>>(Qbf, Kbf, Vtb, out_pb, out_asum, ctxbf);

  // out-proj + residual -> attn_out (f32)
  gemm_kernel<false, false, true><<<g1, 256, 0, stream>>>(ctxbf, fcw_bf, fc_b, src, 1.0f,
                                                          attn_o, nullptr, Dn, Dn, M);

  // LN2 -> FFN
  t5ln_kernel<<<M, 256, 0, stream>>>(attn_o, ln2_w, nxbf, Dn, 1e-6f);
  gemm_kernel<true, false, false><<<dim3(M / GEMM_BM, DFFn / GEMM_BN), 256, 0, stream>>>(
      nxbf, wi_bf, nullptr, nullptr, 1.0f, nullptr, hbf, Dn, DFFn, M);
  gemm_kernel<false, false, true><<<g1, 256, 0, stream>>>(hbf, wo_bf, nullptr, attn_o, 1.0f,
                                                          out_main, nullptr, DFFn, Dn, M);
}
